// RoPEProductKeysEncoderAttention_36043365548424
// MI455X (gfx1250) — matrix operand order repaired and checked by reading, not compiled, not run on hardware
//
#include <hip/hip_runtime.h>

// Problem constants (match reference)
#define B_  2
#define S_  1024
#define DM_ 1024
#define QH_ 16
#define KVH_ 4
#define DH_ 64
#define DHH_ 32
#define M_  32
#define TOPK_ 8

typedef __attribute__((ext_vector_type(16))) __bf16 v16bf;
typedef __attribute__((ext_vector_type(8)))  float  v8f;

union BFrag { v16bf v; uint4 u[2]; };

// LDS byte-offset of a pointer into a __shared__ object (AS3 ptrtoint).
typedef __attribute__((address_space(3))) const void* as3_cvp;
static __device__ __forceinline__ unsigned lds_offset(const void* p) {
  return (unsigned)(size_t)(as3_cvp)p;
}

// ---------------------------------------------------------------------------
// fp32 -> bf16 (round-to-nearest-even)
// ---------------------------------------------------------------------------
__global__ void convert_f32_bf16(const float* __restrict__ src,
                                 unsigned short* __restrict__ dst, int n) {
  int i = blockIdx.x * blockDim.x + threadIdx.x;
  if (i < n) {
    unsigned int u = __float_as_uint(src[i]);
    unsigned int r = u + 0x7FFFu + ((u >> 16) & 1u);
    dst[i] = (unsigned short)(r >> 16);
  }
}

// ---------------------------------------------------------------------------
// C(Mrows x Ncols, fp32) = A(Mrows x K, bf16 rowmajor) * W(Ncols x K, bf16)^T
// 8 waves / block; each wave owns a 16x64 tile (4 x 16x16 f32 accumulators).
// The shared 64x32 B tile is staged into LDS with double-buffered
// GLOBAL_LOAD_ASYNC_TO_LDS_B128 (ASYNCcnt), overlapping the async copy of the
// next K-slab with the WMMAs of the current one.  B-tile column stride is 40
// bf16 (80 B): 16-B aligned for ds_load_b128 and a 20-bank stride so the 16
// lanes of a fragment read 16 distinct LDS banks.
// blockIdx.x covers Mrows/128, blockIdx.y covers Ncols/64.
// ---------------------------------------------------------------------------
#define BT_STRIDE 40            // bf16 elements per staged column
__global__ __launch_bounds__(256)
void gemm_bf16_wmma(const __bf16* __restrict__ A, const __bf16* __restrict__ W,
                    float* __restrict__ C, int K, int Ncols) {
  __shared__ __align__(16) unsigned short btile[2][64 * BT_STRIDE];

  const int tid   = threadIdx.x;
  const int lane  = tid & 31;
  const int wave  = tid >> 5;
  const int rbase = blockIdx.x * 128 + wave * 16;
  const int cbase = blockIdx.y * 64;
  const int lr = lane & 15;   // row (A) / col (B) within the 16x16 tile
  const int hi = lane >> 4;   // lane-half selector

  // Staging slot for this thread: 256 threads x 16 B == one 64x32 bf16 tile.
  const int scol = tid >> 2;            // column 0..63
  const int schk = (tid & 3) * 8;       // bf16 offset 0/8/16/24 within column
  const __bf16* sgp = W + (size_t)(cbase + scol) * K + schk;

  auto stage = [&](int bu, int kb) {
    unsigned loff = lds_offset(&btile[bu][scol * BT_STRIDE + schk]);
    unsigned long long ga = (unsigned long long)(const void*)(sgp + kb);
    asm volatile("global_load_async_to_lds_b128 %0, %1, off"
                 :: "v"(loff), "v"(ga) : "memory");
  };

  const __bf16* arow = A + (size_t)(rbase + lr) * K;

  v8f acc[4];
  acc[0] = v8f{}; acc[1] = v8f{}; acc[2] = v8f{}; acc[3] = v8f{};

  stage(0, 0);                               // prime buffer 0
  for (int kb = 0; kb < K; kb += 32) {
    const int  bu   = (kb >> 5) & 1;
    const bool more = (kb + 32) < K;
    if (more) stage(bu ^ 1, kb + 32);        // async-prefetch next K-slab

    // A fragment (global): lanes 0-15 hold K kb..kb+7 & kb+16..kb+23,
    // lanes 16-31 hold K kb+8..kb+15 & kb+24..kb+31.
    BFrag af;
    af.u[0] = *(const uint4*)(arow + kb + hi * 8);
    af.u[1] = *(const uint4*)(arow + kb + hi * 8 + 16);
    if (more) __builtin_prefetch(arow + kb + 32, 0, 3);

    // Wave's own async writes for buffer `bu` are the oldest outstanding op;
    // loads complete in order, so waiting <=1 retires them while the
    // prefetched slab stays in flight.  Barrier publishes all waves' writes.
    if (more) asm volatile("s_wait_asynccnt 0x1" ::: "memory");
    else      asm volatile("s_wait_asynccnt 0x0" ::: "memory");
    __syncthreads();

#pragma unroll
    for (int t = 0; t < 4; ++t) {
      // B fragment (LDS): lanes 0-15 hold K kb..kb+7 & kb+16..kb+23,
      // lanes 16-31 hold K kb+8..kb+15 & kb+24..kb+31.
      BFrag bf;
      const uint4* lp =
          (const uint4*)&btile[bu][(t * 16 + lr) * BT_STRIDE + hi * 8];
      bf.u[0] = lp[0];
      bf.u[1] = lp[2];
      acc[t] = __builtin_amdgcn_wmma_f32_16x16x32_bf16(
          false, af.v, false, bf.v, (short)0, acc[t], false, false);
    }
    __syncthreads();   // WAR guard before next iteration re-stages bu^1
  }

#pragma unroll
  for (int t = 0; t < 4; ++t) {
    const int col = cbase + t * 16 + lr;
#pragma unroll
    for (int r = 0; r < 8; ++r) {
      const int row = rbase + r + hi * 8;  // lanes 0-15: M 0..7, 16-31: M 8..15
      C[(size_t)row * Ncols + col] = acc[t][r];
    }
  }
}

// ---------------------------------------------------------------------------
// In-place RoPE.  t: (B*S, nheads*64).  One thread per (n, h, i<32) pair.
// ---------------------------------------------------------------------------
__global__ void rope_kernel(float* __restrict__ t, int nheads, int total) {
  int tid = blockIdx.x * blockDim.x + threadIdx.x;
  if (tid >= total) return;
  int i = tid & 31;
  int h = (tid >> 5) % nheads;
  int n = tid / (32 * nheads);
  int s = n & (S_ - 1);
  // inv_freq = ROPE_BASE^(-2i/DH) = exp(-(2i/64)*ln(10000))
  float inv = __expf(-(float)(2 * i) * (9.210340371976184f / 64.0f));
  float ang = (float)s * inv;
  float c = cosf(ang), sn = sinf(ang);
  size_t base = (size_t)n * (nheads * 64) + h * 64 + i;
  float x1 = t[base];
  float x2 = t[base + 32];
  t[base]      = x1 * c - x2 * sn;
  t[base + 32] = x2 * c + x1 * sn;
}

// ---------------------------------------------------------------------------
// k1[b,h,m,:] = sum_j k[b, m*32+j, h, 0:32]
// k2[b,h,m,:] = sum_i k[b, i*32+m, h, 32:64]
// k layout: (B*S, KVH*64).  16384 threads total.
// ---------------------------------------------------------------------------
__global__ void k1k2_kernel(const float* __restrict__ k,
                            float* __restrict__ k1, float* __restrict__ k2) {
  int tid = blockIdx.x * blockDim.x + threadIdx.x;
  if (tid >= 16384) return;
  int which = tid >> 13;
  int r = tid & 8191;
  int d = r & 31;
  int m = (r >> 5) & 31;
  int h = (r >> 10) & (KVH_ - 1);
  int b = r >> 12;
  float acc = 0.f;
  if (which == 0) {
    const float* kp = k + ((size_t)(b * S_ + m * 32) << 8) + h * 64 + d;
#pragma unroll 4
    for (int j = 0; j < 32; ++j) acc += kp[(size_t)j << 8];
    k1[((b * KVH_ + h) * 32 + m) * 32 + d] = acc;
  } else {
    const float* kp = k + ((size_t)(b * S_ + m) << 8) + h * 64 + 32 + d;
#pragma unroll 4
    for (int i = 0; i < 32; ++i) acc += kp[(size_t)i << 13];
    k2[((b * KVH_ + h) * 32 + m) * 32 + d] = acc;
  }
}

// ---------------------------------------------------------------------------
// Selection + attention.  One query per lane; 256-thread block shares (b,h).
// Exploits:  q . concat(vecs1[i], vecs2[j]) == topv1[i] + topv2[j].
// ---------------------------------------------------------------------------
__global__ __launch_bounds__(256)
void select_attn_kernel(const float* __restrict__ q, const float* __restrict__ v,
                        const float* __restrict__ k1, const float* __restrict__ k2,
                        float* __restrict__ ao) {
  __shared__ float k1s[32 * 32];
  __shared__ float k2s[32 * 32];
  const int qidx = blockIdx.x * 256 + threadIdx.x;
  const int s = qidx & (S_ - 1);
  const int h = (qidx >> 10) & (QH_ - 1);
  const int b = qidx >> 14;
  const int kvh = h >> 2;  // n_rep = 4, jnp.repeat semantics

  const float* k1g = k1 + ((b * KVH_ + kvh) << 10);
  const float* k2g = k2 + ((b * KVH_ + kvh) << 10);
  for (int i = threadIdx.x; i < 1024; i += 256) { k1s[i] = k1g[i]; k2s[i] = k2g[i]; }
  __syncthreads();

  const float* qp = q + ((size_t)(b * S_ + s) << 10) + h * 64;

  float qh_[32];
  float tv1[8], tv2[8];
  int   ti1[8], ti2[8];
#pragma unroll
  for (int t = 0; t < 8; ++t) { tv1[t] = -3.4e38f; ti1[t] = 0; tv2[t] = -3.4e38f; ti2[t] = 0; }

  // ---- s1 = q1 . k1[m], sorted-descending top-8 (strict > => jax tie order)
#pragma unroll
  for (int d = 0; d < 32; ++d) qh_[d] = qp[d];
  for (int m = 0; m < 32; ++m) {
    float acc = 0.f;
#pragma unroll
    for (int d = 0; d < 32; ++d) acc += qh_[d] * k1s[m * 32 + d];
    int idx = m;
#pragma unroll
    for (int t = 0; t < 8; ++t) {
      if (acc > tv1[t]) { float tf = tv1[t]; tv1[t] = acc; acc = tf;
                          int   ii = ti1[t]; ti1[t] = idx; idx = ii; }
    }
  }
  // ---- s2 = q2 . k2[m]
#pragma unroll
  for (int d = 0; d < 32; ++d) qh_[d] = qp[32 + d];
  for (int m = 0; m < 32; ++m) {
    float acc = 0.f;
#pragma unroll
    for (int d = 0; d < 32; ++d) acc += qh_[d] * k2s[m * 32 + d];
    int idx = m;
#pragma unroll
    for (int t = 0; t < 8; ++t) {
      if (acc > tv2[t]) { float tf = tv2[t]; tv2[t] = acc; acc = tf;
                          int   ii = ti2[t]; ti2[t] = idx; idx = ii; }
    }
  }

  // ---- top-8 of the 64 pair sums (candidate index c = i*8 + j, ascending)
  float tv3[8]; int ti3[8];
#pragma unroll
  for (int t = 0; t < 8; ++t) { tv3[t] = -3.4e38f; ti3[t] = 0; }
#pragma unroll
  for (int i = 0; i < 8; ++i) {
#pragma unroll
    for (int j = 0; j < 8; ++j) {
      float val = tv1[i] + tv2[j];
      int c = i * 8 + j;
#pragma unroll
      for (int t = 0; t < 8; ++t) {
        if (val > tv3[t]) { float tf = tv3[t]; tv3[t] = val; val = tf;
                            int   ii = ti3[t]; ti3[t] = c;   c   = ii; }
      }
    }
  }

  // ---- softmax over 8 (scale 1/sqrt(64) = 0.125); tv3[0] is the max
  float w[8];
  float mx = tv3[0] * 0.125f;
  float sum = 0.f;
#pragma unroll
  for (int t = 0; t < 8; ++t) { w[t] = __expf(tv3[t] * 0.125f - mx); sum += w[t]; }
  float isum = 1.0f / sum;

  // ---- gather 8 v rows, weighted sum
  float out[64];
#pragma unroll
  for (int d = 0; d < 64; ++d) out[d] = 0.f;
#pragma unroll
  for (int t = 0; t < 8; ++t) {
    int c = ti3[t];
    int vrow = ti1[c >> 3] * 32 + ti2[c & 7];
    const float* vp = v + ((size_t)(b * S_ + vrow) << 8) + kvh * 64;
    float wt = w[t] * isum;
#pragma unroll
    for (int d = 0; d < 64; d += 4) {
      float4 vv = *(const float4*)(vp + d);
      out[d]     += wt * vv.x;
      out[d + 1] += wt * vv.y;
      out[d + 2] += wt * vv.z;
      out[d + 3] += wt * vv.w;
    }
  }

  float* op = ao + ((size_t)(b * S_ + s) << 10) + h * 64;
#pragma unroll
  for (int d = 0; d < 64; d += 4)
    *(float4*)(op + d) = make_float4(out[d], out[d + 1], out[d + 2], out[d + 3]);
}

// ---------------------------------------------------------------------------
// Host-side orchestration
// ---------------------------------------------------------------------------
extern "C" void kernel_launch(void* const* d_in, const int* in_sizes, int n_in,
                              void* d_out, int out_size, void* d_ws, size_t ws_size,
                              hipStream_t stream) {
  const float* x  = (const float*)d_in[0];
  const float* wq = (const float*)d_in[1];
  const float* wk = (const float*)d_in[2];
  const float* wv = (const float*)d_in[3];
  const float* wo = (const float*)d_in[4];
  float* out = (float*)d_out;

  const int NROW = B_ * S_;          // 2048
  char* ws = (char*)d_ws;
  size_t off = 0;
  auto carve = [&](size_t bytes) -> void* {
    void* p = ws + off;
    off = (off + bytes + 255) & ~(size_t)255;
    return p;
  };

  unsigned short* xb  = (unsigned short*)carve((size_t)NROW * DM_ * 2);
  unsigned short* wqb = (unsigned short*)carve((size_t)DM_ * DM_ * 2);
  unsigned short* wkb = (unsigned short*)carve((size_t)(KVH_ * DH_) * DM_ * 2);
  unsigned short* wvb = (unsigned short*)carve((size_t)(KVH_ * DH_) * DM_ * 2);
  unsigned short* wob = (unsigned short*)carve((size_t)DM_ * DM_ * 2);
  float* qf  = (float*)carve((size_t)NROW * DM_ * 4);
  float* kf  = (float*)carve((size_t)NROW * (KVH_ * DH_) * 4);
  float* vf  = (float*)carve((size_t)NROW * (KVH_ * DH_) * 4);
  float* k1f = (float*)carve((size_t)B_ * KVH_ * M_ * DHH_ * 4);
  float* k2f = (float*)carve((size_t)B_ * KVH_ * M_ * DHH_ * 4);
  float* aof = (float*)carve((size_t)NROW * DM_ * 4);
  unsigned short* aob = (unsigned short*)carve((size_t)NROW * DM_ * 2);
  (void)ws_size; (void)in_sizes; (void)n_in; (void)out_size;

  auto cvt = [&](const float* s, unsigned short* d, int n) {
    convert_f32_bf16<<<(n + 255) / 256, 256, 0, stream>>>(s, d, n);
  };

  // bf16 conversions of inputs
  cvt(x,  xb,  NROW * DM_);
  cvt(wq, wqb, DM_ * DM_);
  cvt(wk, wkb, KVH_ * DH_ * DM_);
  cvt(wv, wvb, KVH_ * DH_ * DM_);
  cvt(wo, wob, DM_ * DM_);

  // q/k/v projections (bf16 WMMA, fp32 accumulate)
  gemm_bf16_wmma<<<dim3(NROW / 128, DM_ / 64), 256, 0, stream>>>(
      (const __bf16*)xb, (const __bf16*)wqb, qf, DM_, DM_);
  gemm_bf16_wmma<<<dim3(NROW / 128, (KVH_ * DH_) / 64), 256, 0, stream>>>(
      (const __bf16*)xb, (const __bf16*)wkb, kf, DM_, KVH_ * DH_);
  gemm_bf16_wmma<<<dim3(NROW / 128, (KVH_ * DH_) / 64), 256, 0, stream>>>(
      (const __bf16*)xb, (const __bf16*)wvb, vf, DM_, KVH_ * DH_);

  // RoPE
  {
    int tq = NROW * QH_ * DHH_;   // 1,048,576
    rope_kernel<<<(tq + 255) / 256, 256, 0, stream>>>(qf, QH_, tq);
    int tk = NROW * KVH_ * DHH_;  // 262,144
    rope_kernel<<<(tk + 255) / 256, 256, 0, stream>>>(kf, KVH_, tk);
  }

  // k1 / k2 reductions
  k1k2_kernel<<<64, 256, 0, stream>>>(kf, k1f, k2f);

  // selection + attention (one query per lane)
  select_attn_kernel<<<(B_ * QH_ * S_) / 256, 256, 0, stream>>>(qf, vf, k1f, k2f, aof);

  // output projection
  cvt(aof, aob, NROW * DM_);
  gemm_bf16_wmma<<<dim3(NROW / 128, DM_ / 64), 256, 0, stream>>>(
      (const __bf16*)aob, (const __bf16*)wob, out, DM_, DM_);
}
